// SpanExtractorWithCRF_31035433681519
// MI455X (gfx1250) — matrix operand order repaired and checked by reading, not compiled, not run on hardware
//
#include <hip/hip_runtime.h>
#include <hip/hip_bf16.h>
#include <math.h>

// Problem constants (reference: L,H,NH,DH,FF,V,T,B,S)
static constexpr int kL  = 12;
static constexpr int kH  = 768;
static constexpr int kNH = 12;
static constexpr int kDH = 64;
static constexpr int kFF = 3072;
static constexpr int kT  = 9;
static constexpr int kB  = 8;
static constexpr int kS  = 512;
static constexpr int kN  = kB * kS;   // 4096 tokens

typedef __bf16 bf16;
typedef __attribute__((ext_vector_type(16))) __bf16 v16bf;
typedef __attribute__((ext_vector_type(8)))  float  v8f;

// ---------------------------------------------------------------------------
// CDNA5 async global->LDS copy (ASYNCcnt-tracked), per-lane 16B.
//   LDS offset from one VGPR, global address from a VGPR pair (no SGPR base).
// ---------------------------------------------------------------------------
__device__ __forceinline__ void async_ld_b128(unsigned ldsOff, const void* g)
{
  asm volatile("global_load_async_to_lds_b128 %0, %1, off"
               :: "v"(ldsOff), "v"(g)
               : "memory");
}

__device__ __forceinline__ void wait_asynccnt0()
{
#if __has_builtin(__builtin_amdgcn_s_wait_asynccnt)
  __builtin_amdgcn_s_wait_asynccnt(0);
#else
  asm volatile("s_wait_asynccnt 0x0" ::: "memory");
#endif
}

// ---------------------------------------------------------------------------
// Embedding gather + LayerNorm (block per token, 256 threads, H=768 -> 3/thr)
// ---------------------------------------------------------------------------
__global__ __launch_bounds__(256) void embed_ln_kernel(
    const int* __restrict__ ids, const int* __restrict__ tts,
    const float* __restrict__ we, const float* __restrict__ pe,
    const float* __restrict__ te,
    const float* __restrict__ g, const float* __restrict__ bb,
    float* __restrict__ x, bf16* __restrict__ xb)
{
  const int tok = blockIdx.x;
  const int s   = tok % kS;
  const int id  = ids[tok];
  const int tt  = tts[tok];
  const float* wr = we + (size_t)id * kH;
  const float* pr = pe + (size_t)s  * kH;
  const float* tr = te + (size_t)tt * kH;

  float vals[3];
  float lsum = 0.f, lsq = 0.f;
#pragma unroll
  for (int j = 0; j < 3; ++j) {
    int i = threadIdx.x + j * 256;
    float v = wr[i] + pr[i] + tr[i];
    vals[j] = v; lsum += v; lsq += v * v;
  }
  __shared__ float rs[256], rq[256];
  rs[threadIdx.x] = lsum; rq[threadIdx.x] = lsq;
  __syncthreads();
  for (int off = 128; off > 0; off >>= 1) {
    if (threadIdx.x < off) {
      rs[threadIdx.x] += rs[threadIdx.x + off];
      rq[threadIdx.x] += rq[threadIdx.x + off];
    }
    __syncthreads();
  }
  const float mean = rs[0] / (float)kH;
  const float var  = rq[0] / (float)kH - mean * mean;
  const float inv  = rsqrtf(var + 1e-12f);
#pragma unroll
  for (int j = 0; j < 3; ++j) {
    int i = threadIdx.x + j * 256;
    float v = (vals[j] - mean) * inv * g[i] + bb[i];
    x [(size_t)tok * kH + i] = v;
    xb[(size_t)tok * kH + i] = (bf16)v;
  }
}

// ---------------------------------------------------------------------------
// Residual add + LayerNorm: x = LN(xin + y) (xin/xout may alias; block-local)
// ---------------------------------------------------------------------------
__global__ __launch_bounds__(256) void residual_ln_kernel(
    const float* __restrict__ xin, const float* __restrict__ y,
    const float* __restrict__ g, const float* __restrict__ bb,
    float* __restrict__ xout, bf16* __restrict__ xb)
{
  const int tok = blockIdx.x;
  float vals[3];
  float lsum = 0.f, lsq = 0.f;
#pragma unroll
  for (int j = 0; j < 3; ++j) {
    int i = threadIdx.x + j * 256;
    float v = xin[(size_t)tok * kH + i] + y[(size_t)tok * kH + i];
    vals[j] = v; lsum += v; lsq += v * v;
  }
  __shared__ float rs[256], rq[256];
  rs[threadIdx.x] = lsum; rq[threadIdx.x] = lsq;
  __syncthreads();
  for (int off = 128; off > 0; off >>= 1) {
    if (threadIdx.x < off) {
      rs[threadIdx.x] += rs[threadIdx.x + off];
      rq[threadIdx.x] += rq[threadIdx.x + off];
    }
    __syncthreads();
  }
  const float mean = rs[0] / (float)kH;
  const float var  = rq[0] / (float)kH - mean * mean;
  const float inv  = rsqrtf(var + 1e-12f);
#pragma unroll
  for (int j = 0; j < 3; ++j) {
    int i = threadIdx.x + j * 256;
    float v = (vals[j] - mean) * inv * g[i] + bb[i];
    xout[(size_t)tok * kH + i] = v;
    xb  [(size_t)tok * kH + i] = (bf16)v;
  }
}

// ---------------------------------------------------------------------------
// Weight convert f32 -> bf16 with K-tile swizzle:
//   dst[((k/32)*M + m)*32 + (k%32)] = (bf16) src[k*M + m]
// so each column's 32 K-values for one K-step are contiguous in memory
// (-> B fragments are two ds_load_b128). Writes fully coalesced.
// ---------------------------------------------------------------------------
__global__ void cvt_swizzle_kernel(const float* __restrict__ src,
                                   bf16* __restrict__ dst, int K, int M)
{
  int idx = blockIdx.x * blockDim.x + threadIdx.x;
  if (idx >= K * M) return;
  int kin  = idx & 31;
  int m    = (idx >> 5) % M;
  int kblk = idx / (32 * M);
  dst[idx] = (bf16)src[(size_t)(kblk * 32 + kin) * M + m];
}

// ---------------------------------------------------------------------------
// Tiled bf16 GEMM via v_wmma_f32_16x16x32_bf16 + async global->LDS staging.
//   C[N x M] = A[N x K](bf16, row-major) @ W (bf16, K-tile swizzled) + bias
// Block: 256 thr = 8 waves as 4x2; block tile 64x64; each wave owns a 16x32
// strip (two accumulators sharing one A fragment). Double-buffered LDS
// (tile t lives in buffer t&1), one barrier + one s_wait_asynccnt per step;
// the async copy for tile t+1 is issued right after the barrier so it can
// overlap the fragment loads + WMMAs of tile t.
// op: 0 = none, 1 = tanh-GELU. outF (f32) / outB (bf16) each optional.
// ---------------------------------------------------------------------------
__global__ __launch_bounds__(256) void gemm_bf16_wmma_kernel(
    const bf16* __restrict__ A, const bf16* __restrict__ Wsw,
    const float* __restrict__ bias,
    float* __restrict__ outF, bf16* __restrict__ outB,
    int K, int M, int op)
{
  // 80-byte row stride: keeps every b128 LDS op 16B-aligned, pads banks.
  __shared__ __attribute__((aligned(16))) bf16 la[2][64][40];   // [row][k]
  __shared__ __attribute__((aligned(16))) bf16 lbs[2][64][40];  // [col][k]

  const int tid  = threadIdx.x;
  const int lane = tid & 31;
  const int wave = tid >> 5;
  const int wr   = wave >> 1;       // 0..3: 16-row strip
  const int wc   = wave & 1;        // 0..1: 32-col strip
  const int rowBase = blockIdx.y * 64;
  const int colBase = blockIdx.x * 64;

  // Fragment addressing.
  // A (16x32): row = lane%16, K = 16*(lane>=16) + e -> 16 contiguous elems
  // (same K order as B).
  const int m_local = wr * 16 + (lane & 15);
  const int kh      = (lane >> 4) * 16;
  // B (32x16): col = lane%16, K = 16*(lane>=16) + e -> 16 contiguous elems
  // in the swizzled [col][k] layout.
  const int kb      = (lane >> 4) * 16;
  const int c0      = wc * 32 + (lane & 15);       // cols of acc0
  const int c1      = c0 + 16;                     // cols of acc1

  // Staging coords: A 64x32 (one b128/thr), B 64x32 (one b128/thr).
  const int ar   = tid >> 2;
  const int aseg = (tid & 3) * 8;
  const int bc   = tid >> 2;          // column 0..63
  const int bseg = (tid & 3) * 8;     // k offset 0,8,16,24

  const int kSteps = K >> 5;
  const bf16* aSrc = A + (size_t)(rowBase + ar) * K + aseg;
  const bf16* bSrc = Wsw + (size_t)(colBase + bc) * 32 + bseg;
  const size_t bStep = (size_t)M * 32;   // per K-step advance in swizzled W

  const unsigned laOff0 = (unsigned)(size_t)&la [0][ar][aseg];
  const unsigned laOff1 = (unsigned)(size_t)&la [1][ar][aseg];
  const unsigned lbOff0 = (unsigned)(size_t)&lbs[0][bc][bseg];
  const unsigned lbOff1 = (unsigned)(size_t)&lbs[1][bc][bseg];

  // Kick off tile 0 into buffer 0.
  async_ld_b128(laOff0, aSrc);
  async_ld_b128(lbOff0, bSrc);

  v8f acc0 = {}, acc1 = {};
  int p = 0;
  for (int ks = 0; ks < kSteps; ++ks) {
    wait_asynccnt0();     // this wave's portions of tile ks are in LDS
    __syncthreads();      // everyone's portions are in LDS

    if (ks + 1 < kSteps) {
      // Safe to overwrite buffer p^1: its last readers (iter ks-1) finished
      // their ds_loads before reaching the barrier above.
      async_ld_b128(p ? laOff0 : laOff1, aSrc + (ks + 1) * 32);
      async_ld_b128(p ? lbOff0 : lbOff1, bSrc + (ks + 1) * bStep);
    }

    v16bf afrag, bfrag0, bfrag1;
    const bf16* pa  = &la [p][m_local][0];
    const bf16* pb0 = &lbs[p][c0][kb];
    const bf16* pb1 = &lbs[p][c1][kb];
#pragma unroll
    for (int e = 0; e < 8; ++e) {
      afrag[e]     = pa[kh + e];
      afrag[e + 8] = pa[8 + kh + e];
    }
#pragma unroll
    for (int e = 0; e < 16; ++e) { bfrag0[e] = pb0[e]; bfrag1[e] = pb1[e]; }

    acc0 = __builtin_amdgcn_wmma_f32_16x16x32_bf16(
        false, afrag, false, bfrag0, (short)0, acc0, false, false);
    acc1 = __builtin_amdgcn_wmma_f32_16x16x32_bf16(
        false, afrag, false, bfrag1, (short)0, acc1, false, false);
    p ^= 1;
  }

  // C/D layout: VGPR r, lanes 0-15 -> M=r, lanes 16-31 -> M=8+r; N=lane%16.
  const int ocol0 = colBase + c0;
  const int ocol1 = colBase + c1;
  const float bv0 = bias ? bias[ocol0] : 0.0f;
  const float bv1 = bias ? bias[ocol1] : 0.0f;
#pragma unroll
  for (int r = 0; r < 8; ++r) {
    int orow = rowBase + wr * 16 + ((lane >> 4) << 3) + r;
    float v0 = acc0[r] + bv0;
    float v1 = acc1[r] + bv1;
    if (op == 1) {  // jax.nn.gelu (approximate/tanh form)
      float u0 = 0.7978845608028654f * (v0 + 0.044715f * v0 * v0 * v0);
      float u1 = 0.7978845608028654f * (v1 + 0.044715f * v1 * v1 * v1);
      v0 = 0.5f * v0 * (1.0f + tanhf(u0));
      v1 = 0.5f * v1 * (1.0f + tanhf(u1));
    }
    size_t base = (size_t)orow * M;
    if (outF) { outF[base + ocol0] = v0; outF[base + ocol1] = v1; }
    if (outB) { outB[base + ocol0] = (bf16)v0; outB[base + ocol1] = (bf16)v1; }
  }
}

// ---------------------------------------------------------------------------
// Fused attention: one wave per (b, h, q) row, online softmax over S keys.
// qkv is [N x 3H] f32 (Q | K | V per token). Writes bf16 ctx [N x H].
// ---------------------------------------------------------------------------
__global__ __launch_bounds__(128) void attention_kernel(
    const float* __restrict__ qkv, const int* __restrict__ amask,
    bf16* __restrict__ ctxb)
{
  const int w    = blockIdx.x * 4 + (threadIdx.x >> 5);
  const int lane = threadIdx.x & 31;
  const int q = w % kS;
  const int h = (w / kS) % kNH;
  const int b = w / (kS * kNH);
  const size_t rowStride = 3 * kH;

  const float* qrow = qkv + (size_t)(b * kS + q) * rowStride + h * kDH;
  const float2 qv = *(const float2*)(qrow + lane * 2);

  const float* kbase = qkv + (size_t)(b * kS) * rowStride + kH     + h * kDH;
  const float* vbase = qkv + (size_t)(b * kS) * rowStride + 2 * kH + h * kDH;

  float mrun = -3.0e38f, lrun = 0.0f;
  float2 accv = {0.0f, 0.0f};

  for (int kk = 0; kk < kS; ++kk) {
    float2 kv = *(const float2*)(kbase + (size_t)kk * rowStride + lane * 2);
    float p = qv.x * kv.x + qv.y * kv.y;
#pragma unroll
    for (int off = 16; off > 0; off >>= 1) p += __shfl_xor(p, off, 32);
    float sc = p * 0.125f /* 1/sqrt(64) */
             + (1.0f - (float)amask[b * kS + kk]) * -1.0e9f;
    float mnew  = fmaxf(mrun, sc);
    float scale = __expf(mrun - mnew);   // 0 on first iteration
    float pr    = __expf(sc - mnew);
    float2 vv = *(const float2*)(vbase + (size_t)kk * rowStride + lane * 2);
    lrun   = lrun * scale + pr;
    accv.x = accv.x * scale + pr * vv.x;
    accv.y = accv.y * scale + pr * vv.y;
    mrun = mnew;
  }
  const float invl = 1.0f / lrun;
  size_t o = (size_t)(b * kS + q) * kH + h * kDH + lane * 2;
  ctxb[o]     = (bf16)(accv.x * invl);
  ctxb[o + 1] = (bf16)(accv.y * invl);
}

// ---------------------------------------------------------------------------
// Emissions: em[N x T] = x[N x H] @ Wt[H x T] + bt   (T=9, tiny GEMV)
// ---------------------------------------------------------------------------
__global__ void emissions_kernel(const float* __restrict__ x,
                                 const float* __restrict__ Wt,
                                 const float* __restrict__ bt,
                                 float* __restrict__ em)
{
  int idx = blockIdx.x * blockDim.x + threadIdx.x;
  if (idx >= kN * kT) return;
  int tok = idx / kT, t = idx % kT;
  const float* xr = x + (size_t)tok * kH;
  float s = bt[t];
  for (int k2 = 0; k2 < kH; ++k2) s += xr[k2] * Wt[(size_t)k2 * kT + t];
  em[idx] = s;
}

// ---------------------------------------------------------------------------
// CRF NLL: numerator (per-batch scan) + forward algorithm (b x t parallel).
// Single block, 128 threads. Writes scalar loss to out[0].
// ---------------------------------------------------------------------------
__global__ __launch_bounds__(128) void crf_kernel(
    const float* __restrict__ em, const int* __restrict__ amask,
    const int* __restrict__ labels,
    const float* __restrict__ cstart, const float* __restrict__ cend,
    const float* __restrict__ ctrans, float* __restrict__ out)
{
  __shared__ float trans[kT][kT];
  __shared__ float alpha[kB][kT];
  __shared__ float numv[kB], denv[kB];
  const int tid = threadIdx.x;
  if (tid < kT * kT) trans[tid / kT][tid % kT] = ctrans[tid];
  __syncthreads();

  // Numerator: one thread per batch element.
  if (tid < kB) {
    const int b = tid;
    int tag0 = labels[b * kS];
    float num = cstart[tag0] + em[(size_t)(b * kS) * kT + tag0];
    int prev = tag0;
    int cnt = (amask[b * kS] != 0) ? 1 : 0;
    for (int s2 = 1; s2 < kS; ++s2) {
      int tg = labels[b * kS + s2];
      float mf = (float)amask[b * kS + s2];
      num += (trans[prev][tg] + em[(size_t)(b * kS + s2) * kT + tg]) * mf;
      cnt += (amask[b * kS + s2] != 0) ? 1 : 0;
      prev = tg;
    }
    num += cend[labels[b * kS + (cnt - 1)]];
    numv[b] = num;
  }

  // Forward algorithm: thread (b, t) with tid = b*16 + t.
  const int b = tid >> 4, t = tid & 15;
  const bool act = (b < kB) && (t < kT);
  if (act) alpha[b][t] = cstart[t] + em[(size_t)(b * kS) * kT + t];
  __syncthreads();
  for (int s2 = 1; s2 < kS; ++s2) {
    float res = 0.0f;
    if (act) {
      float mx = -3.0e38f;
#pragma unroll
      for (int i = 0; i < kT; ++i) mx = fmaxf(mx, alpha[b][i] + trans[i][t]);
      float sum = 0.0f;
#pragma unroll
      for (int i = 0; i < kT; ++i)
        sum += __expf(alpha[b][i] + trans[i][t] - mx);
      float v = mx + __logf(sum) + em[(size_t)(b * kS + s2) * kT + t];
      res = (amask[b * kS + s2] != 0) ? v : alpha[b][t];
    }
    __syncthreads();
    if (act) alpha[b][t] = res;
    __syncthreads();
  }

  if (tid < kB) {
    float mx = -3.0e38f;
    for (int i = 0; i < kT; ++i) mx = fmaxf(mx, alpha[tid][i] + cend[i]);
    float sum = 0.0f;
    for (int i = 0; i < kT; ++i) sum += __expf(alpha[tid][i] + cend[i] - mx);
    denv[tid] = mx + __logf(sum);
  }
  __syncthreads();
  if (tid == 0) {
    float acc = 0.0f;
    for (int b2 = 0; b2 < kB; ++b2) acc += numv[b2] - denv[b2];
    out[0] = -acc / (float)kB;
  }
}

// ---------------------------------------------------------------------------
// Host launch
// ---------------------------------------------------------------------------
extern "C" void kernel_launch(void* const* d_in, const int* in_sizes, int n_in,
                              void* d_out, int out_size, void* d_ws,
                              size_t ws_size, hipStream_t stream)
{
  (void)in_sizes; (void)n_in; (void)out_size; (void)ws_size;
  const int*   input_ids      = (const int*)  d_in[0];
  const int*   attention_mask = (const int*)  d_in[1];
  const int*   token_type_ids = (const int*)  d_in[2];
  const int*   labels         = (const int*)  d_in[3];
  const float* word_emb = (const float*)d_in[4];
  const float* pos_emb  = (const float*)d_in[5];
  const float* type_emb = (const float*)d_in[6];
  const float* emb_ln_s = (const float*)d_in[7];
  const float* emb_ln_b = (const float*)d_in[8];
  const float* Wqkv = (const float*)d_in[9];
  const float* bqkv = (const float*)d_in[10];
  const float* Wo   = (const float*)d_in[11];
  const float* bo   = (const float*)d_in[12];
  const float* ln1_s = (const float*)d_in[13];
  const float* ln1_b = (const float*)d_in[14];
  const float* W1   = (const float*)d_in[15];
  const float* b1   = (const float*)d_in[16];
  const float* W2   = (const float*)d_in[17];
  const float* b2   = (const float*)d_in[18];
  const float* ln2_s = (const float*)d_in[19];
  const float* ln2_b = (const float*)d_in[20];
  const float* Wt = (const float*)d_in[21];
  const float* bt = (const float*)d_in[22];
  const float* crf_start = (const float*)d_in[23];
  const float* crf_end   = (const float*)d_in[24];
  const float* crf_trans = (const float*)d_in[25];
  float* outp = (float*)d_out;

  // Workspace carving (256B aligned); total ~106 MB.
  char* ws = (char*)d_ws;
  auto carve = [&](size_t bytes) -> char* {
    char* p = ws;
    ws += (bytes + 255) & ~(size_t)255;
    return p;
  };
  float* x    = (float*)carve((size_t)kN * kH * sizeof(float));
  bf16*  xb   = (bf16*) carve((size_t)kN * kH * sizeof(bf16));
  float* qkv  = (float*)carve((size_t)kN * 3 * kH * sizeof(float));
  float* tmp2 = (float*)carve((size_t)kN * kH * sizeof(float));
  bf16*  ctxb = (bf16*) carve((size_t)kN * kH * sizeof(bf16));
  bf16*  hb   = (bf16*) carve((size_t)kN * kFF * sizeof(bf16));
  bf16*  wb   = (bf16*) carve((size_t)kH * kFF * sizeof(bf16));
  float* em   = (float*)carve((size_t)kN * kT * sizeof(float));

  embed_ln_kernel<<<kN, 256, 0, stream>>>(
      input_ids, token_type_ids, word_emb, pos_emb, type_emb,
      emb_ln_s, emb_ln_b, x, xb);

  for (int l = 0; l < kL; ++l) {
    int n;
    // QKV projection: [N,H] @ [H,3H]
    n = kH * 3 * kH;
    cvt_swizzle_kernel<<<(n + 255) / 256, 256, 0, stream>>>(
        Wqkv + (size_t)l * n, wb, kH, 3 * kH);
    gemm_bf16_wmma_kernel<<<dim3(3 * kH / 64, kN / 64), 256, 0, stream>>>(
        xb, wb, bqkv + (size_t)l * 3 * kH, qkv, nullptr, kH, 3 * kH, 0);

    // Attention (flash-style, wave per query row)
    attention_kernel<<<kB * kNH * kS / 4, 128, 0, stream>>>(
        qkv, attention_mask, ctxb);

    // Output projection: [N,H] @ [H,H]
    n = kH * kH;
    cvt_swizzle_kernel<<<(n + 255) / 256, 256, 0, stream>>>(
        Wo + (size_t)l * n, wb, kH, kH);
    gemm_bf16_wmma_kernel<<<dim3(kH / 64, kN / 64), 256, 0, stream>>>(
        ctxb, wb, bo + (size_t)l * kH, tmp2, nullptr, kH, kH, 0);
    residual_ln_kernel<<<kN, 256, 0, stream>>>(
        x, tmp2, ln1_s + (size_t)l * kH, ln1_b + (size_t)l * kH, x, xb);

    // FFN up + GELU: [N,H] @ [H,FF] -> bf16 only
    n = kH * kFF;
    cvt_swizzle_kernel<<<(n + 255) / 256, 256, 0, stream>>>(
        W1 + (size_t)l * n, wb, kH, kFF);
    gemm_bf16_wmma_kernel<<<dim3(kFF / 64, kN / 64), 256, 0, stream>>>(
        xb, wb, b1 + (size_t)l * kFF, nullptr, hb, kH, kFF, 1);

    // FFN down: [N,FF] @ [FF,H]
    n = kFF * kH;
    cvt_swizzle_kernel<<<(n + 255) / 256, 256, 0, stream>>>(
        W2 + (size_t)l * n, wb, kFF, kH);
    gemm_bf16_wmma_kernel<<<dim3(kH / 64, kN / 64), 256, 0, stream>>>(
        hb, wb, b2 + (size_t)l * kH, tmp2, nullptr, kFF, kH, 0);
    residual_ln_kernel<<<kN, 256, 0, stream>>>(
        x, tmp2, ln2_s + (size_t)l * kH, ln2_b + (size_t)l * kH, x, xb);
  }

  emissions_kernel<<<(kN * kT + 255) / 256, 256, 0, stream>>>(x, Wt, bt, em);
  crf_kernel<<<1, 128, 0, stream>>>(
      em, attention_mask, labels, crf_start, crf_end, crf_trans, outp);
}
